// SelfAttention_2946347565317
// MI455X (gfx1250) — compile-verified, matrix operand order checked by reading, not run on hardware
//
#include <hip/hip_runtime.h>

// ---------------------------------------------------------------------------
// Fused causal self-attention for MI455X (gfx1250, wave32, WMMA + TDM).
// Stages: fp32->bf16 convert, QKV GEMM (V stored transposed [H][Dh][T]),
// flash attention, output projection. All LDS staging via Tensor Data Mover
// (tensor_load_to_lds) with double buffering; f32 accumulation throughout.
// ---------------------------------------------------------------------------

typedef __attribute__((ext_vector_type(16))) __bf16 v16bf;
typedef __attribute__((ext_vector_type(8)))  float  v8f;
typedef __attribute__((ext_vector_type(4)))  unsigned v4u;
typedef __attribute__((ext_vector_type(8)))  unsigned v8u;

#define T_SEQ 4096
#define C_DIM 768
#define H_NUM 12
#define DH 64

union FragU {
  v16bf v;
  unsigned u32[8];
  unsigned short u16[16];
};

__device__ inline unsigned short f32_to_bf16(float f) {
  unsigned u = __float_as_uint(f);
  u += 0x7FFFu + ((u >> 16) & 1u);   // round-to-nearest-even
  return (unsigned short)(u >> 16);
}

// LDS byte offset of a __shared__ object: generic LDS pointers carry the
// workgroup-relative LDS offset in their low 32 bits (ISA 10.2 aperture rules).
__device__ inline unsigned lds_offset(const void* p) {
  return (unsigned)(unsigned long long)p;
}

// ---------------------------------------------------------------------------
// Tensor Data Mover: 2D bf16 tile load global -> LDS (D# per ISA 8.3/8.4;
// groups 2/3 zero => <=2D). pad_enable inserts LDS row padding:
//   interval = 2^(pad_int_code+1) data DWORDs, amount = (pad_amt_code+1) DWORDs.
// Whole-wave op, issue from one wave; tracked by TENSORcnt.
// ---------------------------------------------------------------------------
__device__ inline void tdm_load_tile_2d(unsigned lds_off, const unsigned short* gsrc,
                                        unsigned tile_cols, unsigned tile_rows,
                                        unsigned tensor_cols, unsigned tensor_rows,
                                        unsigned row_stride_elems,
                                        unsigned pad_int_code, unsigned pad_amt_code) {
  unsigned long long ga = (unsigned long long)gsrc;
  v4u g0;
  g0.x = 1u;                                   // count=1
  g0.y = lds_off;                              // lds_addr
  g0.z = (unsigned)ga;                         // global_addr[31:0]
  g0.w = (unsigned)((ga >> 32) & 0x01FFFFFFull) | (2u << 30);  // addr[56:32] | type=2
  v8u g1;
  g1.s0 = (1u << 16)                           // data_size = 2 bytes
        | (1u << 20)                           // pad_enable
        | (pad_int_code << 22) | (pad_amt_code << 25);
  g1.s1 = (tensor_cols & 0xFFFFu) << 16;
  g1.s2 = ((tensor_cols >> 16) & 0xFFFFu) | ((tensor_rows & 0xFFFFu) << 16);
  g1.s3 = ((tensor_rows >> 16) & 0xFFFFu) | ((tile_cols & 0xFFFFu) << 16);
  g1.s4 = (tile_rows & 0xFFFFu);
  g1.s5 = row_stride_elems;
  g1.s6 = 0u;
  g1.s7 = 0u;
  v4u gz = {0u, 0u, 0u, 0u};
  asm volatile("tensor_load_to_lds %0, %1, %2, %3"
               :: "s"(g0), "s"(g1), "s"(gz), "s"(gz)
               : "memory");
}

// A/B fragment (16x32 bf16) per ISA 7.12.2: lane = row, K pairs packed in
// VGPRs, lanes 16-31 take K offset +8/+24. Row-major source with even stride
// -> aligned b32 pair loads (coalesce to ds_load_b128).
__device__ inline v16bf load_frag_rowpairs(const unsigned short* t, int stride,
                                           int rowBase, int kbase) {
  const int lane = threadIdx.x & 31;
  const int half = lane >> 4;
  const int m = lane & 15;
  FragU f;
#pragma unroll
  for (int g = 0; g < 2; ++g)
#pragma unroll
    for (int p = 0; p < 4; ++p) {
      const int k = kbase + g * 16 + half * 8 + 2 * p;
      f.u32[g * 4 + p] = *(const unsigned*)(t + (long)(rowBase + m) * stride + k);
    }
  return f.v;
}

__device__ inline v8f wmma_bf16(v16bf a, v16bf b, v8f c) {
  return __builtin_amdgcn_wmma_f32_16x16x32_bf16(false, a, false, b,
                                                 (short)0, c, false, false);
}

// ---------------------------------------------------------------------------
__global__ void cvt_bf16_kernel(const float* __restrict__ in,
                                unsigned short* __restrict__ out, int n) {
  for (int i = blockIdx.x * blockDim.x + threadIdx.x; i < n;
       i += gridDim.x * blockDim.x)
    out[i] = f32_to_bf16(in[i]);
}

// ---------------------------------------------------------------------------
// QKV projection: qkv[m,n] = sum_k x[m,k] * w_qkv[n,k].
// Q,K scattered to [H][T][Dh]; V scattered to [H][Dh][T] (transposed) so the
// attention P*V B-operand becomes a contiguous-pair LDS fast path.
// 128 threads (4 waves); block tile 128x64 (wave = 32x64, 8 WMMA / K-chunk);
// K-chunks of 32 double-buffered via TDM (LDS pitch 40 = 16DW + 4DW pad).
// ---------------------------------------------------------------------------
__global__ __launch_bounds__(128) void qkv_gemm_kernel(
    const unsigned short* __restrict__ xb, const unsigned short* __restrict__ wb,
    unsigned short* __restrict__ qo, unsigned short* __restrict__ ko,
    unsigned short* __restrict__ vo) {
  __shared__ alignas(16) unsigned short Xs[2][128 * 40];
  __shared__ alignas(16) unsigned short Ws[2][64 * 40];
  const int m0 = blockIdx.x * 128, n0 = blockIdx.y * 64;
  const int t = threadIdx.x;
  const int wave = t >> 5, lane = t & 31, half = lane >> 4, ln = lane & 15;

  v8f acc[2][4];
#pragma unroll
  for (int a = 0; a < 2; ++a)
#pragma unroll
    for (int j = 0; j < 4; ++j)
#pragma unroll
      for (int r = 0; r < 8; ++r) acc[a][j][r] = 0.0f;

  if (wave == 0)
    tdm_load_tile_2d(lds_offset(&Xs[0][0]), xb + (long)m0 * C_DIM,
                     32, 128, C_DIM, T_SEQ, C_DIM, 3, 3);
  else if (wave == 1)
    tdm_load_tile_2d(lds_offset(&Ws[0][0]), wb + (long)n0 * C_DIM,
                     32, 64, C_DIM, 3 * C_DIM, C_DIM, 3, 3);

  int it = 0;
  for (int k0 = 0; k0 < C_DIM; k0 += 32, ++it) {
    const int p = it & 1;
    __builtin_amdgcn_s_wait_tensorcnt(0);   // current buffer DMA complete
    __syncthreads();                        // visible to all; p^1 readers done
    if (k0 + 32 < C_DIM) {                  // overlap next chunk's DMA
      if (wave == 0)
        tdm_load_tile_2d(lds_offset(&Xs[p ^ 1][0]),
                         xb + (long)m0 * C_DIM + k0 + 32,
                         32, 128, C_DIM, T_SEQ, C_DIM, 3, 3);
      else if (wave == 1)
        tdm_load_tile_2d(lds_offset(&Ws[p ^ 1][0]),
                         wb + (long)n0 * C_DIM + k0 + 32,
                         32, 64, C_DIM, 3 * C_DIM, C_DIM, 3, 3);
    }
    // Hoist all fragments, then burst the WMMAs (one dscnt wait).
    const v16bf a0 = load_frag_rowpairs(Xs[p], 40, wave * 32, 0);
    const v16bf a1 = load_frag_rowpairs(Xs[p], 40, wave * 32 + 16, 0);
    v16bf bf[4];
#pragma unroll
    for (int j = 0; j < 4; ++j)
      bf[j] = load_frag_rowpairs(Ws[p], 40, j * 16, 0);  // B[k,n]=W[n,k]
#pragma unroll
    for (int j = 0; j < 4; ++j) acc[0][j] = wmma_bf16(a0, bf[j], acc[0][j]);
#pragma unroll
    for (int j = 0; j < 4; ++j) acc[1][j] = wmma_bf16(a1, bf[j], acc[1][j]);
  }

#pragma unroll
  for (int j = 0; j < 4; ++j) {
    const int n = n0 + j * 16 + ln;
    const int which = n / C_DIM;        // 0=Q,1=K,2=V
    const int c = n - which * C_DIM;
    const int hh = c >> 6, d = c & 63;
#pragma unroll
    for (int a = 0; a < 2; ++a)
#pragma unroll
      for (int r = 0; r < 8; ++r) {
        const int m = m0 + wave * 32 + a * 16 + r + 8 * half;
        const unsigned short val = f32_to_bf16(acc[a][j][r]);
        if (which == 0)      qo[((long)hh * T_SEQ + m) * DH + d] = val;
        else if (which == 1) ko[((long)hh * T_SEQ + m) * DH + d] = val;
        else                 vo[((long)hh * DH + d) * T_SEQ + m] = val;  // V^T
      }
  }
}

// ---------------------------------------------------------------------------
// Flash attention (causal): block = 64 query rows of one head, 4 waves.
// K tile [key][d] and V^T tile [d][key], both 64x64, double-buffered via TDM
// (LDS pitch 72 = 32DW + 4DW pad). Per key block: S = Q*K^T (8 WMMA), online
// softmax, P staged via LDS, O += P*V (8 WMMA) with running rescale.
// ---------------------------------------------------------------------------
__global__ __launch_bounds__(128) void attn_kernel(
    const unsigned short* __restrict__ qb, const unsigned short* __restrict__ kb,
    const unsigned short* __restrict__ vtb, unsigned short* __restrict__ yb) {
  __shared__ alignas(16) unsigned short Ks[2][64 * 72];
  __shared__ alignas(16) unsigned short Vts[2][64 * 72];
  __shared__ alignas(16) unsigned short Ps[4][16 * 72];
  const int h = blockIdx.y;
  const int q0 = blockIdx.x * 64;
  const int t = threadIdx.x;
  const int wave = t >> 5, lane = t & 31, half = lane >> 4, ln = lane & 15;

  const unsigned short* khead  = kb + (long)h * T_SEQ * DH;   // [key][d]
  const unsigned short* vthead = vtb + (long)h * DH * T_SEQ;  // [d][key]

  // Q fragments for this wave's 16 rows, loaded once straight from global.
  const unsigned short* qhead = qb + ((long)h * T_SEQ + q0 + wave * 16) * DH;
  const v16bf qa0 = load_frag_rowpairs(qhead, DH, 0, 0);
  const v16bf qa1 = load_frag_rowpairs(qhead, DH, 0, 32);

  v8f o[4];
  float m_i[8], l_i[8];
#pragma unroll
  for (int j = 0; j < 4; ++j)
#pragma unroll
    for (int r = 0; r < 8; ++r) o[j][r] = 0.0f;
#pragma unroll
  for (int r = 0; r < 8; ++r) { m_i[r] = -3.0e38f; l_i[r] = 0.0f; }

  if (wave == 0)
    tdm_load_tile_2d(lds_offset(&Ks[0][0]), khead, 64, 64, DH, T_SEQ, DH, 4, 3);
  else if (wave == 1)
    tdm_load_tile_2d(lds_offset(&Vts[0][0]), vthead, 64, 64, T_SEQ, DH, T_SEQ, 4, 3);

  const int last = (int)blockIdx.x;
  for (int kblk = 0; kblk <= last; ++kblk) {
    const int k0 = kblk * 64;
    const int p = kblk & 1;
    __builtin_amdgcn_s_wait_tensorcnt(0);
    __syncthreads();
    if (kblk < last) {                      // overlap next block's DMA
      if (wave == 0)
        tdm_load_tile_2d(lds_offset(&Ks[p ^ 1][0]), khead + (long)(k0 + 64) * DH,
                         64, 64, DH, T_SEQ, DH, 4, 3);
      else if (wave == 1)
        tdm_load_tile_2d(lds_offset(&Vts[p ^ 1][0]), vthead + (k0 + 64),
                         64, 64, T_SEQ, DH, T_SEQ, 4, 3);
    }

    // S = Q * K^T : hoist all 8 B-fragments, then burst 8 WMMAs.
    v16bf kf0[4], kf1[4];
#pragma unroll
    for (int j = 0; j < 4; ++j) {
      kf0[j] = load_frag_rowpairs(Ks[p], 72, j * 16, 0);
      kf1[j] = load_frag_rowpairs(Ks[p], 72, j * 16, 32);
    }
    v8f s[4];
#pragma unroll
    for (int j = 0; j < 4; ++j)
#pragma unroll
      for (int r = 0; r < 8; ++r) s[j][r] = 0.0f;
#pragma unroll
    for (int j = 0; j < 4; ++j) s[j] = wmma_bf16(qa0, kf0[j], s[j]);
#pragma unroll
    for (int j = 0; j < 4; ++j) s[j] = wmma_bf16(qa1, kf1[j], s[j]);

    // Online softmax. C layout: row = r + 8*half, col = ln (per 16-col tile j).
#pragma unroll
    for (int r = 0; r < 8; ++r) {
      const int mrow = q0 + wave * 16 + r + 8 * half;
      float mx = -3.0e38f;
#pragma unroll
      for (int j = 0; j < 4; ++j) {
        float v = s[j][r] * 0.125f;                 // 1/sqrt(64)
        const int ncol = k0 + j * 16 + ln;
        v = (ncol <= mrow) ? v : -3.0e38f;          // causal mask
        s[j][r] = v;
        mx = fmaxf(mx, v);
      }
#pragma unroll
      for (int msk = 1; msk < 16; msk <<= 1)
        mx = fmaxf(mx, __shfl_xor(mx, msk, 32));
      const float mnew = fmaxf(m_i[r], mx);
      const float corr = __expf(m_i[r] - mnew);
      float sum = 0.0f;
#pragma unroll
      for (int j = 0; j < 4; ++j) {
        const float pv = __expf(s[j][r] - mnew);
        s[j][r] = pv;
        sum += pv;
      }
#pragma unroll
      for (int msk = 1; msk < 16; msk <<= 1)
        sum += __shfl_xor(sum, msk, 32);
      l_i[r] = l_i[r] * corr + sum;
      m_i[r] = mnew;
#pragma unroll
      for (int j = 0; j < 4; ++j) o[j][r] *= corr;
    }

    // Stage P (C-layout -> row-major LDS) so it can be re-read as A fragments.
    unsigned short* pw = &Ps[wave][0];
#pragma unroll
    for (int j = 0; j < 4; ++j)
#pragma unroll
      for (int r = 0; r < 8; ++r)
        pw[(r + 8 * half) * 72 + j * 16 + ln] = f32_to_bf16(s[j][r]);
    asm volatile("s_wait_dscnt 0" ::: "memory");  // wave-private store->load

    const v16bf pa0 = load_frag_rowpairs(pw, 72, 0, 0);
    const v16bf pa1 = load_frag_rowpairs(pw, 72, 0, 32);
    // B[k,n] = V[k, d=n] = Vt[n][k]: contiguous pairs in Vt rows.
    v16bf vf0[4], vf1[4];
#pragma unroll
    for (int j = 0; j < 4; ++j) {
      vf0[j] = load_frag_rowpairs(Vts[p], 72, j * 16, 0);
      vf1[j] = load_frag_rowpairs(Vts[p], 72, j * 16, 32);
    }
#pragma unroll
    for (int j = 0; j < 4; ++j) o[j] = wmma_bf16(pa0, vf0[j], o[j]);
#pragma unroll
    for (int j = 0; j < 4; ++j) o[j] = wmma_bf16(pa1, vf1[j], o[j]);
  }

  // y[t, h*64+d] = O / l   (bf16 for the projection GEMM)
#pragma unroll
  for (int j = 0; j < 4; ++j)
#pragma unroll
    for (int r = 0; r < 8; ++r) {
      const int mrow = q0 + wave * 16 + r + 8 * half;
      const float val = o[j][r] / l_i[r];
      yb[(long)mrow * C_DIM + h * DH + j * 16 + ln] = f32_to_bf16(val);
    }
}

// ---------------------------------------------------------------------------
// Output projection: out[m,n] = sum_k y[m,k] * w_proj[n,k] (fp32 out).
// Block tile 128x64 (wave = 32x64), TDM double-buffered K-chunks of 32.
// ---------------------------------------------------------------------------
__global__ __launch_bounds__(128) void proj_gemm_kernel(
    const unsigned short* __restrict__ yb, const unsigned short* __restrict__ wb,
    float* __restrict__ out) {
  __shared__ alignas(16) unsigned short Xs[2][128 * 40];
  __shared__ alignas(16) unsigned short Ws[2][64 * 40];
  const int m0 = blockIdx.x * 128, n0 = blockIdx.y * 64;
  const int t = threadIdx.x;
  const int wave = t >> 5, lane = t & 31, half = lane >> 4, ln = lane & 15;

  v8f acc[2][4];
#pragma unroll
  for (int a = 0; a < 2; ++a)
#pragma unroll
    for (int j = 0; j < 4; ++j)
#pragma unroll
      for (int r = 0; r < 8; ++r) acc[a][j][r] = 0.0f;

  if (wave == 0)
    tdm_load_tile_2d(lds_offset(&Xs[0][0]), yb + (long)m0 * C_DIM,
                     32, 128, C_DIM, T_SEQ, C_DIM, 3, 3);
  else if (wave == 1)
    tdm_load_tile_2d(lds_offset(&Ws[0][0]), wb + (long)n0 * C_DIM,
                     32, 64, C_DIM, C_DIM, C_DIM, 3, 3);

  int it = 0;
  for (int k0 = 0; k0 < C_DIM; k0 += 32, ++it) {
    const int p = it & 1;
    __builtin_amdgcn_s_wait_tensorcnt(0);
    __syncthreads();
    if (k0 + 32 < C_DIM) {
      if (wave == 0)
        tdm_load_tile_2d(lds_offset(&Xs[p ^ 1][0]),
                         yb + (long)m0 * C_DIM + k0 + 32,
                         32, 128, C_DIM, T_SEQ, C_DIM, 3, 3);
      else if (wave == 1)
        tdm_load_tile_2d(lds_offset(&Ws[p ^ 1][0]),
                         wb + (long)n0 * C_DIM + k0 + 32,
                         32, 64, C_DIM, C_DIM, C_DIM, 3, 3);
    }
    const v16bf a0 = load_frag_rowpairs(Xs[p], 40, wave * 32, 0);
    const v16bf a1 = load_frag_rowpairs(Xs[p], 40, wave * 32 + 16, 0);
    v16bf bf[4];
#pragma unroll
    for (int j = 0; j < 4; ++j)
      bf[j] = load_frag_rowpairs(Ws[p], 40, j * 16, 0);
#pragma unroll
    for (int j = 0; j < 4; ++j) acc[0][j] = wmma_bf16(a0, bf[j], acc[0][j]);
#pragma unroll
    for (int j = 0; j < 4; ++j) acc[1][j] = wmma_bf16(a1, bf[j], acc[1][j]);
  }

#pragma unroll
  for (int j = 0; j < 4; ++j) {
    const int n = n0 + j * 16 + ln;
#pragma unroll
    for (int a = 0; a < 2; ++a)
#pragma unroll
      for (int r = 0; r < 8; ++r) {
        const int m = m0 + wave * 32 + a * 16 + r + 8 * half;
        out[(long)m * C_DIM + n] = acc[a][j][r];
      }
  }
}

// ---------------------------------------------------------------------------
extern "C" void kernel_launch(void* const* d_in, const int* in_sizes, int n_in,
                              void* d_out, int out_size, void* d_ws, size_t ws_size,
                              hipStream_t stream) {
  const float* x     = (const float*)d_in[0];  // [1,4096,768]
  const float* wqkv  = (const float*)d_in[1];  // [2304,768]
  const float* wproj = (const float*)d_in[2];  // [768,768]
  // d_in[3] = causal mask (tril) — structure hardcoded in attn_kernel.

  unsigned short* ws = (unsigned short*)d_ws;
  const long NX  = (long)T_SEQ * C_DIM;   // 3,145,728
  const long NWQ = 3L * C_DIM * C_DIM;    // 1,769,472
  const long NWP = (long)C_DIM * C_DIM;   //   589,824

  unsigned short* xb     = ws;
  unsigned short* wqkvb  = xb + NX;
  unsigned short* wprojb = wqkvb + NWQ;
  unsigned short* qbuf   = wprojb + NWP;
  unsigned short* kbuf   = qbuf + NX;
  unsigned short* vtbuf  = kbuf + NX;     // V stored transposed [H][Dh][T]
  unsigned short* ybuf   = xb;            // reuse x-bf16 region after QKV GEMM

  cvt_bf16_kernel<<<1024, 256, 0, stream>>>(x, xb, (int)NX);
  cvt_bf16_kernel<<<1024, 256, 0, stream>>>(wqkv, wqkvb, (int)NWQ);
  cvt_bf16_kernel<<<512, 256, 0, stream>>>(wproj, wprojb, (int)NWP);

  qkv_gemm_kernel<<<dim3(T_SEQ / 128, (3 * C_DIM) / 64), 128, 0, stream>>>(
      xb, wqkvb, qbuf, kbuf, vtbuf);

  attn_kernel<<<dim3(T_SEQ / 64, H_NUM), 128, 0, stream>>>(
      qbuf, kbuf, vtbuf, ybuf);

  proj_gemm_kernel<<<dim3(T_SEQ / 128, C_DIM / 64), 128, 0, stream>>>(
      ybuf, wprojb, (float*)d_out);
}
